// Net_GAT_29824252903778
// MI455X (gfx1250) — matrix operand order repaired and checked by reading, not compiled, not run on hardware
//
#include <hip/hip_runtime.h>
#include <hip/hip_bf16.h>

#define NNODES 20000
#define NEDGES 320000
#define ETOT   (NEDGES + NNODES)
#define HID    64
#define NH     4
#define HC     64      // per-head channels
#define HGC    256     // NH*HC
#define CDIM   8
#define NEG_SLOPE 0.2f

typedef __attribute__((ext_vector_type(16))) __bf16 v16bf;
typedef __attribute__((ext_vector_type(8)))  float  v8f;

struct Pack32 { uint4 lo, hi; };   // 32 bytes == v16bf

__device__ __forceinline__ unsigned short f2bf(float f) {
  unsigned u = __float_as_uint(f);
  u += 0x7FFFu + ((u >> 16) & 1u);           // round-to-nearest-even
  return (unsigned short)(u >> 16);
}
__device__ __forceinline__ unsigned f2mono(float f) {
  unsigned u = __float_as_uint(f);
  return (u & 0x80000000u) ? ~u : (u | 0x80000000u);   // order-preserving map
}
__device__ __forceinline__ float mono2f(unsigned u) {
  return (u & 0x80000000u) ? __uint_as_float(u & 0x7FFFFFFFu) : __uint_as_float(~u);
}
__device__ __forceinline__ float elu(float x) { return x > 0.f ? x : (__expf(x) - 1.f); }

// ---------------------------------------------------------------- edges: int64 -> int32, append self loops
__global__ void prep_edges_kernel(const long long* __restrict__ ei,
                                  int* __restrict__ s, int* __restrict__ d) {
  int t = blockIdx.x * blockDim.x + threadIdx.x;
  if (t >= ETOT) return;
  if (t < NEDGES) { s[t] = (int)ei[t]; d[t] = (int)ei[NEDGES + t]; }
  else            { s[t] = t - NEDGES; d[t] = t - NEDGES; }
}

// ---------------------------------------------------------------- weights: f32 [64,256] -> bf16 WMMA-swizzled
// layout: [layer][nt(16)][ks(2)][lane(32)][i(16)]
//   B element i of lane L: col = nt*16+(L&15), K = ks*32 + (L>>4)*16 + i
__global__ void prep_weights_kernel(const float* __restrict__ W1, const float* __restrict__ W2,
                                    const float* __restrict__ W3, unsigned short* __restrict__ Wswz) {
  int t = blockIdx.x * blockDim.x + threadIdx.x;
  if (t >= 3 * 16 * 2 * 32 * 16) return;
  int i    =  t        & 15;
  int lane = (t >> 4)  & 31;
  int ks   = (t >> 9)  &  1;
  int nt   = (t >> 10) & 15;
  int l    =  t >> 14;
  int n = nt * 16 + (lane & 15);
  int k = ks * 32 + (lane >> 4) * 16 + i;
  const float* W = (l == 0) ? W1 : (l == 1 ? W2 : W3);
  Wswz[t] = f2bf(W[k * HGC + n]);
}

// ---------------------------------------------------------------- encoder MLP 8->32->64 (ELU), per node
__global__ void encoder_kernel(const float* __restrict__ x,
                               const float* __restrict__ W1, const float* __restrict__ b1,
                               const float* __restrict__ W2, const float* __restrict__ b2,
                               float* __restrict__ hcur, unsigned short* __restrict__ hb) {
  __shared__ float sW1[8 * 32], sb1[32], sW2[32 * 64], sb2[64];
  for (int i = threadIdx.x; i < 8 * 32;  i += blockDim.x) sW1[i] = W1[i];
  for (int i = threadIdx.x; i < 32;      i += blockDim.x) sb1[i] = b1[i];
  for (int i = threadIdx.x; i < 32 * 64; i += blockDim.x) sW2[i] = W2[i];
  for (int i = threadIdx.x; i < 64;      i += blockDim.x) sb2[i] = b2[i];
  __syncthreads();
  int n = blockIdx.x * blockDim.x + threadIdx.x;
  if (n >= NNODES) return;
  float xi[8];
#pragma unroll
  for (int i = 0; i < 8; ++i) xi[i] = x[n * 8 + i];
  float t1[32];
  for (int j = 0; j < 32; ++j) {
    float a = sb1[j];
#pragma unroll
    for (int i = 0; i < 8; ++i) a += xi[i] * sW1[i * 32 + j];
    t1[j] = elu(a);
  }
  for (int j = 0; j < 64; ++j) {
    float a = sb2[j];
#pragma unroll 8
    for (int i = 0; i < 32; ++i) a += t1[i] * sW2[i * 64 + j];
    a = elu(a);
    hcur[(size_t)n * HID + j] = a;
    hb  [(size_t)n * HID + j] = f2bf(a);
  }
}

// ---------------------------------------------------------------- WMMA GEMM: hg[N,256] = hb[N,64] @ W[64,256]
// one wave per 16-row M-tile: A tile loaded once into 16 VGPRs, then 16 N-tiles
// x 2 k-steps = 32 v_wmma_f32_16x16x32_bf16 per wave with full A register reuse.
__global__ void gemm_bf16_wmma_kernel(const unsigned short* __restrict__ Abf,   // [M,64] bf16 row-major
                                      const unsigned short* __restrict__ Bswz,  // swizzled weights
                                      float* __restrict__ C, int M) {
  int mt   = (blockIdx.x * blockDim.x + threadIdx.x) >> 5;   // wave id == m-tile
  int lane = threadIdx.x & 31;
  if (mt * 16 >= M) return;    // uniform across wave

  int r    = lane & 15;
  int half = lane >> 4;
  int koff = half * 16;

  // A layout per k-step: elem i -> K=ks*32+koff+i
  v16bf a[2];
#pragma unroll
  for (int ks = 0; ks < 2; ++ks) {
    const unsigned short* ap = Abf + (size_t)(mt * 16 + r) * HID + ks * 32 + koff;
    Pack32 au;
    au.lo = *(const uint4*)(ap);
    au.hi = *(const uint4*)(ap + 8);
    a[ks] = __builtin_bit_cast(v16bf, au);
  }

  int mbase = mt * 16 + half * 8;
#pragma unroll
  for (int nt = 0; nt < 16; ++nt) {
    v8f acc = {};
#pragma unroll
    for (int ks = 0; ks < 2; ++ks) {
      const unsigned short* bp = Bswz + (size_t)(((nt * 2 + ks) * 32 + lane) * 16);
      Pack32 bu;
      bu.lo = *(const uint4*)(bp);
      bu.hi = *(const uint4*)(bp + 8);
      v16bf b = __builtin_bit_cast(v16bf, bu);
      acc = __builtin_amdgcn_wmma_f32_16x16x32_bf16(false, a[ks], false, b, (short)0, acc,
                                                    false, false);
    }
    // D layout: lane col = nt*16+(L&15); VGPR v -> row mbase + v
    float* cp = C + (size_t)mbase * HGC + nt * 16 + r;
#pragma unroll
    for (int v = 0; v < 8; ++v)
      cp[(size_t)v * HGC] = acc[v];
  }
}

// ---------------------------------------------------------------- per-node attention dots + zero accumulators
__global__ void node_att_kernel(const float* __restrict__ hg,
                                const float* __restrict__ att_s, const float* __restrict__ att_d,
                                float* __restrict__ asrc, float* __restrict__ adst,
                                unsigned* __restrict__ mmax, float* __restrict__ denom,
                                float* __restrict__ hout) {
  int t = blockIdx.x * blockDim.x + threadIdx.x;
  if (t >= NNODES * NH) return;
  int n = t >> 2, h = t & 3;
  const float* row = hg + (size_t)n * HGC + h * HC;
  float s = 0.f, d = 0.f;
#pragma unroll 8
  for (int c = 0; c < HC; ++c) {
    float v = row[c];
    s += v * att_s[h * HC + c];
    d += v * att_d[h * HC + c];
  }
  asrc[t] = s; adst[t] = d;
  mmax[t] = 0u;            // mono-key "-inf"
  denom[t] = 0.f;
  float* ho = hout + (size_t)n * HID + h * 16;
#pragma unroll
  for (int j = 0; j < 16; ++j) ho[j] = 0.f;
}

// ---------------------------------------------------------------- edge pass 1: logits + segment-max
__global__ void edge_logits_max_kernel(const int* __restrict__ src, const int* __restrict__ dst,
                                       const float* __restrict__ asrc, const float* __restrict__ adst,
                                       float* __restrict__ elog, unsigned* __restrict__ mmax) {
  int e = blockIdx.x * blockDim.x + threadIdx.x;
  if (e >= ETOT) return;
  int s = src[e], d = dst[e];
#pragma unroll
  for (int h = 0; h < NH; ++h) {
    float l = asrc[s * NH + h] + adst[d * NH + h];
    l = l > 0.f ? l : NEG_SLOPE * l;
    elog[(size_t)e * NH + h] = l;
    atomicMax(&mmax[d * NH + h], f2mono(l));
  }
}

// ---------------------------------------------------------------- edge pass 2: exp + segment-sum
__global__ void edge_expsum_kernel(const int* __restrict__ dst, float* __restrict__ elog,
                                   const unsigned* __restrict__ mmax, float* __restrict__ denom) {
  int e = blockIdx.x * blockDim.x + threadIdx.x;
  if (e >= ETOT) return;
  int d = dst[e];
#pragma unroll
  for (int h = 0; h < NH; ++h) {
    float m  = mono2f(mmax[d * NH + h]);
    float ex = __expf(elog[(size_t)e * NH + h] - m);
    elog[(size_t)e * NH + h] = ex;
    atomicAdd(&denom[d * NH + h], ex);
  }
}

// ---------------------------------------------------------------- edge pass 3: weighted aggregate (mean folded)
// one wave per edge; lane = channel c handles c and c+32
__global__ void edge_aggregate_kernel(const int* __restrict__ src, const int* __restrict__ dst,
                                      const float* __restrict__ elog, const float* __restrict__ denom,
                                      const float* __restrict__ hg, float* __restrict__ hout) {
  int gid  = blockIdx.x * blockDim.x + threadIdx.x;
  int e    = gid >> 5;
  int lane = threadIdx.x & 31;
  if (e >= ETOT) return;
  int s = src[e], d = dst[e];
  const float* hrow = hg + (size_t)s * HGC;
  __builtin_prefetch(hrow, 0, 1);
  float w[NH];
#pragma unroll
  for (int h = 0; h < NH; ++h)
    w[h] = elog[(size_t)e * NH + h] / (denom[d * NH + h] + 1e-16f) * (1.0f / NH);
  float m0 = 0.f, m1 = 0.f;
#pragma unroll
  for (int h = 0; h < NH; ++h) {
    m0 += hrow[h * HC + lane]      * w[h];
    m1 += hrow[h * HC + lane + 32] * w[h];
  }
  atomicAdd(&hout[(size_t)d * HID + lane],      m0);
  atomicAdd(&hout[(size_t)d * HID + lane + 32], m1);
}

// ---------------------------------------------------------------- residual + bias + bf16 recompress
__global__ void residual_kernel(const float* __restrict__ hout, const float* __restrict__ bias,
                                float* __restrict__ hcur, unsigned short* __restrict__ hb) {
  int t = blockIdx.x * blockDim.x + threadIdx.x;
  if (t >= NNODES * HID) return;
  int c = t & (HID - 1);
  float v = hout[t] + hcur[t] + bias[c];
  hcur[t] = v;
  hb[t]   = f2bf(v);
}

// ---------------------------------------------------------------- output MLP 64->64->32->8 (ELU,ELU,linear)
__global__ void out_mlp_kernel(const float* __restrict__ hcur,
                               const float* __restrict__ W1, const float* __restrict__ b1,
                               const float* __restrict__ W2, const float* __restrict__ b2,
                               const float* __restrict__ W3, const float* __restrict__ b3,
                               float* __restrict__ out) {
  __shared__ float sW1[64 * 64], sW2[64 * 32], sW3[32 * 8], sb1[64], sb2[32], sb3[8];
  for (int i = threadIdx.x; i < 64 * 64; i += blockDim.x) sW1[i] = W1[i];
  for (int i = threadIdx.x; i < 64 * 32; i += blockDim.x) sW2[i] = W2[i];
  for (int i = threadIdx.x; i < 32 * 8;  i += blockDim.x) sW3[i] = W3[i];
  for (int i = threadIdx.x; i < 64;      i += blockDim.x) sb1[i] = b1[i];
  for (int i = threadIdx.x; i < 32;      i += blockDim.x) sb2[i] = b2[i];
  for (int i = threadIdx.x; i < 8;       i += blockDim.x) sb3[i] = b3[i];
  __syncthreads();
  int n = blockIdx.x * blockDim.x + threadIdx.x;
  if (n >= NNODES) return;
  float xi[64];
#pragma unroll 8
  for (int i = 0; i < 64; ++i) xi[i] = hcur[(size_t)n * HID + i];
  float t1[64];
  for (int j = 0; j < 64; ++j) {
    float a = sb1[j];
#pragma unroll 8
    for (int i = 0; i < 64; ++i) a += xi[i] * sW1[i * 64 + j];
    t1[j] = elu(a);
  }
  float t2[32];
  for (int j = 0; j < 32; ++j) {
    float a = sb2[j];
#pragma unroll 8
    for (int i = 0; i < 64; ++i) a += t1[i] * sW2[i * 32 + j];
    t2[j] = elu(a);
  }
#pragma unroll
  for (int k = 0; k < CDIM; ++k) {
    float a = sb3[k];
#pragma unroll 8
    for (int i = 0; i < 32; ++i) a += t2[i] * sW3[i * CDIM + k];
    out[(size_t)n * CDIM + k] = a;
  }
}

// ----------------------------------------------------------------
extern "C" void kernel_launch(void* const* d_in, const int* in_sizes, int n_in,
                              void* d_out, int out_size, void* d_ws, size_t ws_size,
                              hipStream_t stream) {
  const float*     x      = (const float*)d_in[0];
  const long long* ei     = (const long long*)d_in[1];
  const float*     W_enc1 = (const float*)d_in[2];
  const float*     b_enc1 = (const float*)d_in[3];
  const float*     W_enc2 = (const float*)d_in[4];
  const float*     b_enc2 = (const float*)d_in[5];
  const float* Wg[3] = {(const float*)d_in[6],  (const float*)d_in[10], (const float*)d_in[14]};
  const float* As[3] = {(const float*)d_in[7],  (const float*)d_in[11], (const float*)d_in[15]};
  const float* Ad[3] = {(const float*)d_in[8],  (const float*)d_in[12], (const float*)d_in[16]};
  const float* Bg[3] = {(const float*)d_in[9],  (const float*)d_in[13], (const float*)d_in[17]};
  const float* W_o1 = (const float*)d_in[18]; const float* b_o1 = (const float*)d_in[19];
  const float* W_o2 = (const float*)d_in[20]; const float* b_o2 = (const float*)d_in[21];
  const float* W_o3 = (const float*)d_in[22]; const float* b_o3 = (const float*)d_in[23];
  float* out = (float*)d_out;

  uintptr_t base = (uintptr_t)d_ws;
  size_t off = 0;
  auto carve = [&](size_t bytes) -> void* {
    void* p = (void*)(base + off);
    off += (bytes + 255) & ~(size_t)255;
    return p;
  };
  int*            srcI  = (int*)carve(sizeof(int) * (size_t)ETOT);
  int*            dstI  = (int*)carve(sizeof(int) * (size_t)ETOT);
  unsigned short* hb    = (unsigned short*)carve(sizeof(unsigned short) * (size_t)NNODES * HID);
  unsigned short* Wswz  = (unsigned short*)carve(sizeof(unsigned short) * 3 * 16 * 2 * 32 * 16);
  float*          hcur  = (float*)carve(sizeof(float) * (size_t)NNODES * HID);
  float*          hg    = (float*)carve(sizeof(float) * (size_t)NNODES * HGC);
  float*          asrc  = (float*)carve(sizeof(float) * (size_t)NNODES * NH);
  float*          adst  = (float*)carve(sizeof(float) * (size_t)NNODES * NH);
  unsigned*       mmax  = (unsigned*)carve(sizeof(unsigned) * (size_t)NNODES * NH);
  float*          denom = (float*)carve(sizeof(float) * (size_t)NNODES * NH);
  float*          elog  = (float*)carve(sizeof(float) * (size_t)ETOT * NH);
  float*          hout  = (float*)carve(sizeof(float) * (size_t)NNODES * HID);

  prep_edges_kernel<<<(ETOT + 255) / 256, 256, 0, stream>>>(ei, srcI, dstI);
  prep_weights_kernel<<<(3 * 16384 + 255) / 256, 256, 0, stream>>>(Wg[0], Wg[1], Wg[2], Wswz);
  encoder_kernel<<<(NNODES + 255) / 256, 256, 0, stream>>>(x, W_enc1, b_enc1, W_enc2, b_enc2,
                                                           hcur, hb);
  const int MTILES = (NNODES + 15) / 16;                 // 1250 waves, 8 waves/block
  for (int l = 0; l < 3; ++l) {
    gemm_bf16_wmma_kernel<<<(MTILES + 7) / 8, 256, 0, stream>>>(hb, Wswz + l * 16384, hg, NNODES);
    node_att_kernel<<<(NNODES * NH + 255) / 256, 256, 0, stream>>>(hg, As[l], Ad[l], asrc, adst,
                                                                   mmax, denom, hout);
    edge_logits_max_kernel<<<(ETOT + 255) / 256, 256, 0, stream>>>(srcI, dstI, asrc, adst,
                                                                   elog, mmax);
    edge_expsum_kernel<<<(ETOT + 255) / 256, 256, 0, stream>>>(dstI, elog, mmax, denom);
    edge_aggregate_kernel<<<(ETOT * 32 + 255) / 256, 256, 0, stream>>>(srcI, dstI, elog, denom,
                                                                       hg, hout);
    residual_kernel<<<(NNODES * HID + 255) / 256, 256, 0, stream>>>(hout, Bg[l], hcur, hb);
  }
  out_mlp_kernel<<<(NNODES + 127) / 128, 128, 0, stream>>>(hcur, W_o1, b_o1, W_o2, b_o2,
                                                           W_o3, b_o3, out);
}
